// CausalSelfAttention_49632642073155
// MI455X (gfx1250) — matrix operand order repaired and checked by reading, not compiled, not run on hardware
//
#include <hip/hip_runtime.h>
#include <hip/hip_bf16.h>
#include <math.h>

// ---- WMMA vector types (gfx1250, wave32) ----
typedef __attribute__((ext_vector_type(16))) __bf16 v16bf;
typedef __attribute__((ext_vector_type(8)))  __bf16 v8bf;
typedef __attribute__((ext_vector_type(4)))  __bf16 v4bf;
typedef __attribute__((ext_vector_type(8)))  float  v8f;

// ---- TDM descriptor vector types (6-arg toolchain: (u32x4, i32x8, i32x4, i32x4, i32x8, i32)) ----
typedef unsigned int v4ui __attribute__((ext_vector_type(4)));
typedef int          v8si __attribute__((ext_vector_type(8)));
typedef int          v4si __attribute__((ext_vector_type(4)));

#if __has_builtin(__builtin_amdgcn_tensor_load_to_lds)
#define HAVE_TDM 1
#else
#define HAVE_TDM 0
#endif

#define WMMA_BF16(a, b, c) \
    __builtin_amdgcn_wmma_f32_16x16x32_bf16(false, (a), false, (b), (short)0, (c), false, false)

// Problem constants
#define BB   2
#define LL   2048
#define DD   1024
#define HH   16
#define KH   64          // head dim
#define LD3  3072        // 3*D row stride of qkv

// ---------------------------------------------------------------------------
// f32 -> bf16 conversion (bandwidth-trivial, one pass)
// ---------------------------------------------------------------------------
__global__ void cvt_f32_bf16(const float* __restrict__ src, __bf16* __restrict__ dst, int n) {
    int idx = (blockIdx.x * blockDim.x + threadIdx.x) * 4;
    if (idx + 3 < n) {
        float4 f = *(const float4*)(src + idx);
        v4bf o;
        o[0] = (__bf16)f.x; o[1] = (__bf16)f.y; o[2] = (__bf16)f.z; o[3] = (__bf16)f.w;
        *(v4bf*)(dst + idx) = o;
    } else {
        for (int i = idx; i < n; ++i) dst[i] = (__bf16)src[i];
    }
}

// ---------------------------------------------------------------------------
// bf16 GEMM: out[M,N] = A[M,Kd] @ Wt[N,Kd]^T + bias[N]
// Wave: 16(M) x 64(N) via 4 f32 accumulators. Block: 8 waves -> 64 x 128 tile.
// ---------------------------------------------------------------------------
template <bool OUT_F32>
__global__ void gemm_bf16(const __bf16* __restrict__ A, const __bf16* __restrict__ Wt,
                          const float* __restrict__ bias, void* __restrict__ out,
                          int M, int N, int Kd) {
    const int lane = threadIdx.x & 31;
    const int w    = threadIdx.x >> 5;
    const int half = lane >> 4;
    const int l16  = lane & 15;
    const int rowBase = blockIdx.x * 64 + (w & 3) * 16;
    const int colBase = blockIdx.y * 128 + (w >> 2) * 64;

    v8f acc[4] = {};
    const __bf16* arow = A + (size_t)(rowBase + l16) * Kd;

    for (int kd = 0; kd < Kd; kd += 32) {
        __builtin_prefetch(arow + kd + 256, 0, 1);   // global_prefetch_b8
        v8bf lo = *(const v8bf*)(arow + kd + half * 8);
        v8bf hi = *(const v8bf*)(arow + kd + 16 + half * 8);
        v16bf af = __builtin_shufflevector(lo, hi, 0,1,2,3,4,5,6,7,8,9,10,11,12,13,14,15);
#pragma unroll
        for (int nt = 0; nt < 4; ++nt) {
            const __bf16* wrow = Wt + (size_t)(colBase + nt * 16 + l16) * Kd + kd + half * 16;
            v16bf bf = __builtin_shufflevector(*(const v8bf*)(wrow - half * 8), *(const v8bf*)(wrow + 16 - half * 8), 0,1,2,3,4,5,6,7,8,9,10,11,12,13,14,15);
            acc[nt] = WMMA_BF16(af, bf, acc[nt]);
        }
    }

#pragma unroll
    for (int nt = 0; nt < 4; ++nt) {
        const int col = colBase + nt * 16 + l16;
        const float bv = bias[col];
#pragma unroll
        for (int r = 0; r < 8; ++r) {
            const int row = rowBase + r + 8 * half;  // C-layout: M = vgpr + 8*(lane>=16)
            const float val = acc[nt][r] + bv;
            if (OUT_F32)
                ((float*)out)[(size_t)row * N + col] = val;
            else
                ((__bf16*)out)[(size_t)row * N + col] = (__bf16)val;
        }
    }
}

// ---------------------------------------------------------------------------
// TDM: 2D tile load Global->LDS, 32 rows x 64 bf16 cols, row stride LD3 elems.
// D# layout per CDNA5 ISA. Issued by one wave per WG.
// ---------------------------------------------------------------------------
#if HAVE_TDM
__device__ __forceinline__ void tdm_load_k_tile(unsigned lds_off, unsigned long long gaddr) {
    v4ui g0;
    g0[0] = 1u;                                            // count=1, user mode
    g0[1] = lds_off;                                       // lds_addr (bytes)
    g0[2] = (unsigned)(gaddr & 0xffffffffull);             // global_addr[31:0]
    g0[3] = (unsigned)((gaddr >> 32) & 0x1ffffffull)       // global_addr[56:32]
          | (2u << 30);                                    // type=2 ("image")
    v8si g1;
    g1[0] = (int)(1u << 16);          // workgroup_mask=0, data_size=1 (2 bytes)
    g1[1] = (int)(64u << 16);         // tensor_dim0[15:0]=64
    g1[2] = (int)(2048u << 16);       // tensor_dim0[31:16]=0, tensor_dim1[15:0]=2048
    g1[3] = (int)(64u << 16);         // tensor_dim1[31:16]=0, tile_dim0=64
    g1[4] = 32;                       // tile_dim1=32, tile_dim2=0
    g1[5] = LD3;                      // tensor_dim0_stride[31:0]=3072
    g1[6] = 0;                        // stride[47:32]=0, tensor_dim1_stride[15:0]=0
    g1[7] = 0;
    v4si z4 = {};
    v8si z8 = {};
    __builtin_amdgcn_tensor_load_to_lds(g0, g1, z4, z4, z8, 0);
}
#endif

// ---------------------------------------------------------------------------
// Flash attention. Grid (L/128, B*H), block 256 (8 waves).
// K tile staged by TDM (tensor_load_to_lds, TENSORcnt); V staged transposed by
// VALU path. Causal: fully-masked key blocks skip all compute; elementwise
// mask only on the diagonal band. 1/sqrt(64) folded into Q fragments (exact).
// ---------------------------------------------------------------------------
__global__ void attn_kernel(const __bf16* __restrict__ qkv, __bf16* __restrict__ ctx) {
    __shared__ __bf16 Kt[32][64];       // [key][headdim]
    __shared__ __bf16 Vt[64][32];       // transposed: [headdim][key]
    __shared__ __bf16 Pt[8][16][32];    // per-wave P tile

    const int lane = threadIdx.x & 31;
    const int w    = threadIdx.x >> 5;
    const int half = lane >> 4;
    const int l16  = lane & 15;

    const int b = blockIdx.y >> 4;
    const int h = blockIdx.y & 15;

    const __bf16* qb  = qkv + (size_t)b * LL * LD3 + h * KH;
    const __bf16* kbp = qb + DD;
    const __bf16* vbp = qb + 2 * DD;

    const int qrow0 = blockIdx.x * 128 + w * 16;

    // Q fragments for kdim [0,32) and [32,64), pre-scaled by 1/sqrt(64)=0.125
    v16bf qf[2];
    {
        const __bf16* qr = qb + (size_t)(qrow0 + l16) * LD3;
#pragma unroll
        for (int kt = 0; kt < 2; ++kt) {
            v8bf lo = *(const v8bf*)(qr + kt * 32 + half * 8);
            v8bf hi = *(const v8bf*)(qr + kt * 32 + 16 + half * 8);
            v16bf q = __builtin_shufflevector(lo, hi, 0,1,2,3,4,5,6,7,8,9,10,11,12,13,14,15);
#pragma unroll
            for (int i = 0; i < 16; ++i) q[i] = q[i] * (__bf16)0.125f;  // exact (pow2)
            qf[kt] = q;
        }
    }

    float m_i[8], l_i[8];
#pragma unroll
    for (int r = 0; r < 8; ++r) { m_i[r] = -3.0e38f; l_i[r] = 0.0f; }
    v8f oacc[4] = {};

    const int nkb = (blockIdx.x + 1) * 4;   // key blocks covering this WG's rows
    for (int kbi = 0; kbi < nkb; ++kbi) {
        const int kbase = kbi * 32;
        __syncthreads();   // previous tile fully consumed

#if HAVE_TDM
        if (w == 0) {      // one TDM descriptor per workgroup for the K tile
            tdm_load_k_tile((unsigned)(uintptr_t)&Kt[0][0],
                            (unsigned long long)(uintptr_t)(kbp + (size_t)kbase * LD3));
        }
#else
        {
            const int idx = threadIdx.x * 8;
            const int row = idx >> 6, col = idx & 63;
            *(v8bf*)&Kt[row][col] = *(const v8bf*)(kbp + (size_t)(kbase + row) * LD3 + col);
        }
#endif
        {   // V tile, transposed into LDS (TDM cannot transpose)
            const int idx = threadIdx.x * 8;
            const int row = idx >> 6, col = idx & 63;
            v8bf vv = *(const v8bf*)(vbp + (size_t)(kbase + row) * LD3 + col);
#pragma unroll
            for (int i = 0; i < 8; ++i) Vt[col + i][row] = vv[i];
        }
#if HAVE_TDM
        if (w == 0) __builtin_amdgcn_s_wait_tensorcnt(0);
#endif
        __syncthreads();

        if (kbase <= qrow0 + 15) {   // skip key blocks entirely above our rows
            // S = Q @ K^T (16 x 32)
            v8f st[2] = {};
#pragma unroll
            for (int nt = 0; nt < 2; ++nt)
#pragma unroll
                for (int kt = 0; kt < 2; ++kt) {
                    v16bf bk = __builtin_shufflevector(*(const v8bf*)&Kt[nt * 16 + l16][kt * 32 + half * 8], *(const v8bf*)&Kt[nt * 16 + l16][kt * 32 + 16 + half * 8], 0,1,2,3,4,5,6,7,8,9,10,11,12,13,14,15);
                    st[nt] = WMMA_BF16(qf[kt], bk, st[nt]);
                }

            // causal mask only on the diagonal band
            if (kbase + 31 > qrow0) {
#pragma unroll
                for (int nt = 0; nt < 2; ++nt) {
                    const int col = kbase + nt * 16 + l16;
#pragma unroll
                    for (int r = 0; r < 8; ++r) {
                        const int row = qrow0 + r + 8 * half;
                        st[nt][r] = (col <= row) ? st[nt][r] : -3.0e38f;
                    }
                }
            }

            // online softmax per row (rows live in lane-groups of 16)
#pragma unroll
            for (int r = 0; r < 8; ++r) {
                float mx = fmaxf(st[0][r], st[1][r]);
                mx = fmaxf(mx, __shfl_xor(mx, 1, 32));
                mx = fmaxf(mx, __shfl_xor(mx, 2, 32));
                mx = fmaxf(mx, __shfl_xor(mx, 4, 32));
                mx = fmaxf(mx, __shfl_xor(mx, 8, 32));
                const float mn = fmaxf(m_i[r], mx);
                const float sc = __expf(m_i[r] - mn);
                const float p0 = __expf(st[0][r] - mn);
                const float p1 = __expf(st[1][r] - mn);
                float rs = p0 + p1;
                rs += __shfl_xor(rs, 1, 32);
                rs += __shfl_xor(rs, 2, 32);
                rs += __shfl_xor(rs, 4, 32);
                rs += __shfl_xor(rs, 8, 32);
                l_i[r] = l_i[r] * sc + rs;
                m_i[r] = mn;
#pragma unroll
                for (int nt = 0; nt < 4; ++nt) oacc[nt][r] *= sc;
                // spill P in C layout (per-wave tile; same-wave DS ops in order)
                Pt[w][r + 8 * half][l16]      = (__bf16)p0;
                Pt[w][r + 8 * half][16 + l16] = (__bf16)p1;
            }

            // reload P in A layout
            v8bf plo = *(const v8bf*)&Pt[w][l16][half * 8];
            v8bf phi = *(const v8bf*)&Pt[w][l16][16 + half * 8];
            v16bf pf = __builtin_shufflevector(plo, phi, 0,1,2,3,4,5,6,7,8,9,10,11,12,13,14,15);

            // O += P @ V
#pragma unroll
            for (int nt = 0; nt < 4; ++nt) {
                v16bf bv = __builtin_shufflevector(*(const v8bf*)&Vt[nt * 16 + l16][half * 8], *(const v8bf*)&Vt[nt * 16 + l16][16 + half * 8], 0,1,2,3,4,5,6,7,8,9,10,11,12,13,14,15);
                oacc[nt] = WMMA_BF16(pf, bv, oacc[nt]);
            }
        }
    }

    // epilogue: normalize and store bf16 context
#pragma unroll
    for (int nt = 0; nt < 4; ++nt) {
        const int col = h * KH + nt * 16 + l16;
#pragma unroll
        for (int r = 0; r < 8; ++r) {
            const int row = qrow0 + r + 8 * half;
            const float o = oacc[nt][r] / l_i[r];
            ctx[((size_t)b * LL + row) * DD + col] = (__bf16)o;
        }
    }
}

// ---------------------------------------------------------------------------
extern "C" void kernel_launch(void* const* d_in, const int* in_sizes, int n_in,
                              void* d_out, int out_size, void* d_ws, size_t ws_size,
                              hipStream_t stream) {
    const float* x  = (const float*)d_in[0];   // [B,L,D]
    const float* Wa = (const float*)d_in[1];   // [3D,D]
    const float* ba = (const float*)d_in[2];   // [3D]
    const float* Wp = (const float*)d_in[3];   // [D,D]
    const float* bp = (const float*)d_in[4];   // [D]

    constexpr int nX  = BB * LL * DD;        // 4,194,304
    constexpr int nWa = 3 * DD * DD;         // 3,145,728
    constexpr int nWp = DD * DD;             // 1,048,576

    char* ws = (char*)d_ws;
    __bf16* Xbf  = (__bf16*)(ws);                               //  8,388,608 B
    __bf16* Wabf = (__bf16*)(ws + 8388608);                     //  6,291,456 B
    __bf16* Wpbf = (__bf16*)(ws + 14680064);                    //  2,097,152 B
    __bf16* QKV  = (__bf16*)(ws + 16777216);                    // 25,165,824 B
    __bf16* CTX  = (__bf16*)(ws + 41943040);                    //  8,388,608 B

    // 1) down-convert once to bf16
    cvt_f32_bf16<<<(nX / 4 + 255) / 256, 256, 0, stream>>>(x, Xbf, nX);
    cvt_f32_bf16<<<(nWa / 4 + 255) / 256, 256, 0, stream>>>(Wa, Wabf, nWa);
    cvt_f32_bf16<<<(nWp / 4 + 255) / 256, 256, 0, stream>>>(Wp, Wpbf, nWp);

    // 2) QKV = X @ W_attn^T + b_attn   (M=4096, N=3072, K=1024)
    gemm_bf16<false><<<dim3(4096 / 64, 3072 / 128), 256, 0, stream>>>(
        Xbf, Wabf, ba, (void*)QKV, BB * LL, 3 * DD, DD);

    // 3) flash attention -> CTX (bf16 [B*L, D])
    attn_kernel<<<dim3(LL / 128, BB * HH), 256, 0, stream>>>(QKV, CTX);

    // 4) out = CTX @ W_proj^T + b_proj  (f32 to d_out)
    gemm_bf16<true><<<dim3(4096 / 64, 1024 / 128), 256, 0, stream>>>(
        CTX, Wpbf, bp, d_out, BB * LL, DD, DD);
}
